// GraphEdgeActionGNN_8693013807715
// MI455X (gfx1250) — matrix operand order repaired and checked by reading, not compiled, not run on hardware
//
#include <hip/hip_runtime.h>
#include <hip/hip_bf16.h>

typedef __attribute__((ext_vector_type(16))) _Float16 v16h;
typedef __attribute__((ext_vector_type(8)))  _Float16 v8h;
typedef __attribute__((ext_vector_type(4)))  _Float16 v4h;
typedef __attribute__((ext_vector_type(8)))  float    v8f;
typedef __attribute__((ext_vector_type(4)))  float    v4f;
typedef __attribute__((ext_vector_type(4)))  int      v4i;

#define NB 2048          // graphs
#define NN 64            // nodes per graph
#define HD 128           // hidden
#define NE 512           // edges per graph (NN * DEG)
#define ETOT (NB * NE)   // total edges
#define NL 3
#define WSZ (HD * HD)    // one weight matrix, in halves

union V16H { v16h v; v8h h8[2]; _Float16 h[16]; };

// CDNA5 async global->LDS copy path (ASYNCcnt), guarded.
#if defined(__has_builtin)
#if __has_builtin(__builtin_amdgcn_global_load_async_to_lds_b128) && \
    __has_builtin(__builtin_amdgcn_s_wait_asynccnt)
#define HAVE_ASYNC_LDS 1
#endif
#endif
#ifndef HAVE_ASYNC_LDS
#define HAVE_ASYNC_LDS 0
#endif

#if HAVE_ASYNC_LDS
typedef __attribute__((address_space(1))) v4i* gv4i_p;  // global
typedef __attribute__((address_space(3))) v4i* lv4i_p;  // LDS
#endif

// ---------------------------------------------------------------------------
// Prep: convert 12 layer weight matrices (128x128 fp32, [K][N]) into f16
// transposed [N][K] so WMMA B-fragments are contiguous per lane.
// Slot order: gin_W1[0..2], gin_W2[0..2], post_W1[0..2], post_W2[0..2]
// ---------------------------------------------------------------------------
__global__ __launch_bounds__(256) void prep_weights(
    const float* __restrict__ gW1, const float* __restrict__ gW2,
    const float* __restrict__ pW1, const float* __restrict__ pW2,
    _Float16* __restrict__ Wt)
{
    int idx = blockIdx.x * 256 + threadIdx.x;
    if (idx >= 12 * WSZ) return;
    int m = idx >> 14;          // matrix slot 0..11
    int r = idx & 16383;
    int n = r >> 7;
    int k = r & 127;
    int l = m % 3, grp = m / 3;
    const float* src;
    if      (grp == 0) src = gW1 + l * WSZ;
    else if (grp == 1) src = gW2 + l * WSZ;
    else if (grp == 2) src = pW1 + l * WSZ;
    else               src = pW2 + l * WSZ;
    Wt[m * WSZ + n * HD + k] = (_Float16)src[k * HD + n];
}

// ---------------------------------------------------------------------------
// Weight staging: global f16 [N][K] matrix -> LDS, async if available.
// 8 x b128 per thread => 8 async ops outstanding per wave per staging.
// ---------------------------------------------------------------------------
__device__ __forceinline__ void stage_weights(const _Float16* __restrict__ Wg,
                                              _Float16* __restrict__ wl, int tid)
{
#if HAVE_ASYNC_LDS
#pragma unroll
    for (int i = 0; i < 8; ++i) {
        int off = (i * 256 + tid) * 8;       // halves, 16B per op
        __builtin_amdgcn_global_load_async_to_lds_b128(
            (gv4i_p)(Wg + off), (lv4i_p)(wl + off), 0, 0);
    }
#else
#pragma unroll
    for (int i = 0; i < 8; ++i) {
        int off = i * 256 + tid;
        ((v8h*)wl)[off] = ((const v8h*)Wg)[off];
    }
#endif
}

template <int N>
__device__ __forceinline__ void wait_stage()
{
#if HAVE_ASYNC_LDS
    __builtin_amdgcn_s_wait_asynccnt(N);
#endif
}

// ---------------------------------------------------------------------------
// WMMA fragment loads (CDNA5 layouts, wave32)
// ---------------------------------------------------------------------------
// A 16x32 f16: lanes 0-15 row M=lane, K={k0..k0+7, k0+16..k0+23};
// lanes 16-31 same rows, K={k0+8..k0+15, k0+24..k0+31}.
__device__ __forceinline__ v16h load_afrag(const _Float16* __restrict__ A,
                                           int mt, int k0, int lane)
{
    int row = mt * 16 + (lane & 15);
    int kb  = k0 + ((lane >> 4) << 3);
    V16H a;
    a.h8[0] = *(const v8h*)(A + row * HD + kb);
    a.h8[1] = *(const v8h*)(A + row * HD + kb + 16);
    return a.v;
}

// B 32x16 f16 from [N][K]: lane<16 -> N=lane, K={k0..k0+7, k0+16..k0+23};
// lane>=16 -> N=lane-16, K={k0+8..k0+15, k0+24..k0+31}.
__device__ __forceinline__ v16h load_bfrag(const _Float16* __restrict__ W,
                                           int nt, int k0, int lane)
{
    int n  = nt * 16 + (lane & 15);
    int kk = k0 + ((lane >> 4) << 3);
    V16H b;
    b.h8[0] = *(const v8h*)(W + n * HD + kk);
    b.h8[1] = *(const v8h*)(W + n * HD + kk + 16);
    return b.v;
}

// D = A(64x128 LDS f16) @ W(128x128 f16 [N][K], LDS) + bias -> D(LDS f32)
// Per wave: 1 M-tile x 4 N-tiles; A fragment loaded once per k-step.
__device__ __forceinline__ void mm_64x128x128(const _Float16* __restrict__ A,
                                              const _Float16* __restrict__ W,
                                              const float* __restrict__ bias,
                                              float* __restrict__ D,
                                              int wave, int lane)
{
    const int mt  = wave >> 1;
    const int ntb = (wave & 1) * 4;
    v8f acc[4] = {{}, {}, {}, {}};
#pragma unroll
    for (int kc = 0; kc < 4; ++kc) {
        const int k0 = kc * 32;
        v16h af = load_afrag(A, mt, k0, lane);
#pragma unroll
        for (int j = 0; j < 4; ++j) {
            v16h bf = load_bfrag(W, ntb + j, k0, lane);
            acc[j] = __builtin_amdgcn_wmma_f32_16x16x32_f16(
                false, af, false, bf, (short)0, acc[j], false, false);
        }
    }
    // C/D layout: VGPR r -> M = mt*16 + r + 8*(lane>=16), N = nt*16 + (lane&15)
    const int mb = mt * 16 + ((lane >> 4) << 3);
#pragma unroll
    for (int j = 0; j < 4; ++j) {
        int cn = (ntb + j) * 16 + (lane & 15);
        float bv = bias[cn];
#pragma unroll
        for (int r = 0; r < 8; ++r)
            D[(mb + r) * HD + cn] = acc[j][r] + bv;
    }
}

// Per-node LayerNorm over H=128, one wave per node, in place.
__device__ __forceinline__ void layernorm_rows(float* __restrict__ buf,
                                               const float* __restrict__ g,
                                               const float* __restrict__ b,
                                               int wave, int lane)
{
    for (int n = wave; n < NN; n += 8) {
        float* row = buf + n * HD;
        float s = 0.f;
#pragma unroll
        for (int i = 0; i < 4; ++i) s += row[lane + 32 * i];
#pragma unroll
        for (int off = 16; off > 0; off >>= 1) s += __shfl_xor(s, off, 32);
        float mu = s * (1.f / 128.f);
        float v = 0.f;
#pragma unroll
        for (int i = 0; i < 4; ++i) {
            float d = row[lane + 32 * i] - mu;
            v += d * d;
        }
#pragma unroll
        for (int off = 16; off > 0; off >>= 1) v += __shfl_xor(v, off, 32);
        float rs = rsqrtf(v * (1.f / 128.f) + 1e-5f);
#pragma unroll
        for (int i = 0; i < 4; ++i) {
            int c = lane + 32 * i;
            row[c] = (row[c] - mu) * rs * g[c] + b[c];
        }
    }
}

// Vectorized f32 -> f16 conversion (optional relu).
template <bool RELU>
__device__ __forceinline__ void cvt_f16(const float* __restrict__ s,
                                        _Float16* __restrict__ d, int tid)
{
#pragma unroll
    for (int i = tid; i < NN * HD / 4; i += 256) {
        v4f v = ((const v4f*)s)[i];
        if (RELU) {
            v.x = fmaxf(v.x, 0.f); v.y = fmaxf(v.y, 0.f);
            v.z = fmaxf(v.z, 0.f); v.w = fmaxf(v.w, 0.f);
        }
        v4h h = { (_Float16)v.x, (_Float16)v.y, (_Float16)v.z, (_Float16)v.w };
        ((v4h*)d)[i] = h;
    }
}

// ---------------------------------------------------------------------------
// Main kernel: one block (256 threads, 8 waves) per graph.
// ---------------------------------------------------------------------------
__global__ __launch_bounds__(256) void gnn_graph_kernel(
    const int*   __restrict__ node_features,
    const int*   __restrict__ edge_index,   // [2][ETOT]
    const float* __restrict__ emb,          // [64][128]
    const _Float16* __restrict__ Wt,        // 12 x [128][128] f16 transposed
    const float* __restrict__ gin_b1, const float* __restrict__ gin_ln_g,
    const float* __restrict__ gin_ln_b, const float* __restrict__ gin_b2,
    const float* __restrict__ post_b1, const float* __restrict__ post_b2,
    const float* __restrict__ norm_g, const float* __restrict__ norm_b,
    const float* __restrict__ exit_W1, const float* __restrict__ exit_b1,
    const float* __restrict__ exit_ln_g, const float* __restrict__ exit_ln_b,
    const float* __restrict__ exit_W2, const float* __restrict__ exit_b2,
    float* __restrict__ out)
{
    extern __shared__ char smem[];
    float*    xb  = (float*)smem;                 // 64x128 f32 (32 KB)
    float*    hb  = xb + NN * HD;                 // 64x128 f32 (32 KB), reused as gram
    _Float16* hf  = (_Float16*)(hb + NN * HD);    // 64x128 f16 (16 KB)
    _Float16* wb0 = hf + NN * HD;                 // weight buf 0 (32 KB)
    _Float16* wb1 = wb0 + WSZ;                    // weight buf 1 (32 KB)
    float*    aux = (float*)(wb1 + WSZ);          // 256 f32

    const int g    = blockIdx.x;
    const int tid  = threadIdx.x;
    const int lane = tid & 31;
    const int wave = tid >> 5;

    // Prefetch first weight matrix (gin_W1[0]) into wb0 during the gather.
    stage_weights(Wt + 0 * WSZ, wb0, tid);

    // ---- gather embeddings into xb ----
    const int* nf = node_features + g * NN;
    for (int i = tid; i < NN * HD; i += 256) {
        int n = i >> 7, f = i & 127;
        xb[i] = emb[nf[n] * HD + f];
    }
    __syncthreads();

    const int* srcp = edge_index + g * NE;
    const int* dstp = edge_index + ETOT + g * NE;
    const int  base = g * NN;

    for (int l = 0; l < NL; ++l) {
        // ---- h = x + scatter_add(x[src] -> dst) ----
        for (int i = tid; i < NN * HD / 4; i += 256)
            ((v4f*)hb)[i] = ((const v4f*)xb)[i];
        __syncthreads();
        for (int w = tid; w < NE * 32; w += 256) {
            int e = w >> 5;
            int f = (w & 31) << 2;
            int s = srcp[e] - base;
            int d = dstp[e] - base;
            v4f vx = *(const v4f*)(xb + s * HD + f);
            atomicAdd(hb + d * HD + f + 0, vx.x);
            atomicAdd(hb + d * HD + f + 1, vx.y);
            atomicAdd(hb + d * HD + f + 2, vx.z);
            atomicAdd(hb + d * HD + f + 3, vx.w);
        }
        __syncthreads();

        // ---- h1 = LN(h @ gin_W1 + b1) ----  (wb0; prefetch gin_W2 -> wb1)
        cvt_f16<false>(hb, hf, tid);
        stage_weights(Wt + (3 + l) * WSZ, wb1, tid);
        wait_stage<8>();
        __syncthreads();
        mm_64x128x128(hf, wb0, gin_b1 + l * HD, hb, wave, lane);
        __syncthreads();
        layernorm_rows(hb, gin_ln_g + l * HD, gin_ln_b + l * HD, wave, lane);
        __syncthreads();

        // ---- h2 = relu(h1) @ gin_W2 + b2 ----  (wb1; prefetch post_W1 -> wb0)
        cvt_f16<true>(hb, hf, tid);
        stage_weights(Wt + (6 + l) * WSZ, wb0, tid);
        wait_stage<8>();
        __syncthreads();
        mm_64x128x128(hf, wb1, gin_b2 + l * HD, hb, wave, lane);
        __syncthreads();

        // ---- h3a = h2 @ post_W1 + pb1 ----  (wb0; prefetch post_W2 -> wb1)
        cvt_f16<false>(hb, hf, tid);
        stage_weights(Wt + (9 + l) * WSZ, wb1, tid);
        wait_stage<8>();
        __syncthreads();
        mm_64x128x128(hf, wb0, post_b1 + l * HD, hb, wave, lane);
        __syncthreads();

        // ---- h3 = relu(h3a) @ post_W2 + pb2 ----  (wb1; prefetch next gin_W1)
        cvt_f16<true>(hb, hf, tid);
        if (l + 1 < NL) {
            stage_weights(Wt + (l + 1) * WSZ, wb0, tid);
            wait_stage<8>();
        } else {
            wait_stage<0>();
        }
        __syncthreads();
        mm_64x128x128(hf, wb1, post_b2 + l * HD, hb, wave, lane);
        __syncthreads();

        // ---- residual + global LN into xb ----
        for (int i = tid; i < NN * HD / 4; i += 256) {
            v4f h = ((const v4f*)hb)[i];
            if (l > 0) {
                v4f x = ((const v4f*)xb)[i];
                h.x += x.x; h.y += x.y; h.z += x.z; h.w += x.w;
            }
            ((v4f*)xb)[i] = h;
        }
        __syncthreads();
        layernorm_rows(xb, norm_g, norm_b, wave, lane);
        __syncthreads();
    }

    // =========== exit head: mean -> MLP -> LN -> relu -> scalar ===========
    float* mvec = aux;        // 128
    float* evec = aux + 128;  // 128
    if (tid < HD) {
        float s = 0.f;
        for (int n = 0; n < NN; ++n) s += xb[n * HD + tid];
        mvec[tid] = s * (1.f / (float)NN);
    }
    __syncthreads();
    if (tid < HD) {
        float s = 0.f;
        for (int k = 0; k < HD; ++k) s += mvec[k] * exit_W1[k * HD + tid];
        evec[tid] = s + exit_b1[tid];
    }
    __syncthreads();
    if (wave == 0) {
        float s = evec[lane] + evec[lane + 32] + evec[lane + 64] + evec[lane + 96];
#pragma unroll
        for (int off = 16; off > 0; off >>= 1) s += __shfl_xor(s, off, 32);
        float mu = s * (1.f / 128.f);
        float v = 0.f;
#pragma unroll
        for (int i = 0; i < 4; ++i) {
            float d = evec[lane + 32 * i] - mu;
            v += d * d;
        }
#pragma unroll
        for (int off = 16; off > 0; off >>= 1) v += __shfl_xor(v, off, 32);
        float rs = rsqrtf(v * (1.f / 128.f) + 1e-5f);
        float dot = 0.f;
#pragma unroll
        for (int i = 0; i < 4; ++i) {
            int c = lane + 32 * i;
            float e = fmaxf((evec[c] - mu) * rs * exit_ln_g[c] + exit_ln_b[c], 0.f);
            dot += e * exit_W2[c];
        }
#pragma unroll
        for (int off = 16; off > 0; off >>= 1) dot += __shfl_xor(dot, off, 32);
        if (lane == 0) out[g * 2017 + 2016] = dot + exit_b2[0];
    }

    // =========== Gram matrix: x @ x^T / sqrt(H), upper triangle ===========
    cvt_f16<false>(xb, hf, tid);
    __syncthreads();

    float* gram = hb;  // reuse as 64x64
    {
        const int mt  = wave >> 1;            // 4x4 tiles, 2 per wave
        const int ntb = (wave & 1) * 2;
        v8f acc[2] = {{}, {}};
#pragma unroll
        for (int kc = 0; kc < 4; ++kc) {
            const int k0 = kc * 32;
            v16h af = load_afrag(hf, mt, k0, lane);
#pragma unroll
            for (int j = 0; j < 2; ++j) {
                v16h bf = load_bfrag(hf, ntb + j, k0, lane);  // B col n = x row n
                acc[j] = __builtin_amdgcn_wmma_f32_16x16x32_f16(
                    false, af, false, bf, (short)0, acc[j], false, false);
            }
        }
        const int mb = mt * 16 + ((lane >> 4) << 3);
#pragma unroll
        for (int j = 0; j < 2; ++j) {
            int cn = (ntb + j) * 16 + (lane & 15);
#pragma unroll
            for (int r = 0; r < 8; ++r)
                gram[(mb + r) * NN + cn] = acc[j][r] * 0.08838834764831845f;
        }
    }
    __syncthreads();

    float* orow = out + (size_t)g * 2017;
    for (int idx = tid; idx < NN * NN; idx += 256) {
        int i = idx >> 6, j = idx & 63;
        if (j > i) {
            int t = i * 63 - (i * (i - 1)) / 2 + (j - i - 1);
            orow[t] = gram[idx];
        }
    }
}

// ---------------------------------------------------------------------------
extern "C" void kernel_launch(void* const* d_in, const int* in_sizes, int n_in,
                              void* d_out, int out_size, void* d_ws, size_t ws_size,
                              hipStream_t stream) {
    (void)in_sizes; (void)n_in; (void)out_size; (void)ws_size;
    const int*   node_features = (const int*)  d_in[0];
    const int*   edge_index    = (const int*)  d_in[1];
    const float* emb           = (const float*)d_in[3];
    const float* gin_W1        = (const float*)d_in[4];
    const float* gin_b1        = (const float*)d_in[5];
    const float* gin_ln_g      = (const float*)d_in[6];
    const float* gin_ln_b      = (const float*)d_in[7];
    const float* gin_W2        = (const float*)d_in[8];
    const float* gin_b2        = (const float*)d_in[9];
    const float* post_W1       = (const float*)d_in[10];
    const float* post_b1       = (const float*)d_in[11];
    const float* post_W2       = (const float*)d_in[12];
    const float* post_b2       = (const float*)d_in[13];
    const float* norm_g        = (const float*)d_in[14];
    const float* norm_b        = (const float*)d_in[15];
    const float* exit_W1       = (const float*)d_in[16];
    const float* exit_b1       = (const float*)d_in[17];
    const float* exit_ln_g     = (const float*)d_in[18];
    const float* exit_ln_b     = (const float*)d_in[19];
    const float* exit_W2       = (const float*)d_in[20];
    const float* exit_b2       = (const float*)d_in[21];
    float* out = (float*)d_out;
    _Float16* Wt = (_Float16*)d_ws;

    prep_weights<<<(12 * WSZ + 255) / 256, 256, 0, stream>>>(
        gin_W1, gin_W2, post_W1, post_W2, Wt);

    size_t shmem = (size_t)(NN * HD) * 4 * 2   // xb + hb
                 + (size_t)(NN * HD) * 2       // hf
                 + (size_t)WSZ * 2 * 2         // wb0 + wb1
                 + 256 * 4;                    // aux
    gnn_graph_kernel<<<NB, 256, shmem, stream>>>(
        node_features, edge_index, emb, Wt,
        gin_b1, gin_ln_g, gin_ln_b, gin_b2, post_b1, post_b2,
        norm_g, norm_b, exit_W1, exit_b1, exit_ln_g, exit_ln_b,
        exit_W2, exit_b2, out);
}
